// GATModule_13743895347735
// MI455X (gfx1250) — matrix operand order repaired and checked by reading, not compiled, not run on hardware
//
#include <hip/hip_runtime.h>
#include <math.h>

// ---------------- model constants ----------------
#define IN_CH   64
#define HEADS   8
#define CPH     16
#define HC      128     // HEADS*CPH
#define EDIM    3
#define OUTC    32
#define NSLOPE  0.2f

typedef __attribute__((ext_vector_type(16))) __bf16 bf16x16;
typedef __attribute__((ext_vector_type(8)))  float  f32x8;

// ---- monotonic uint key for float atomic-max (init key 0 acts as -inf) ----
__device__ __forceinline__ unsigned fkey(float f) {
    unsigned b = __float_as_uint(f);
    return (b & 0x80000000u) ? ~b : (b | 0x80000000u);
}
__device__ __forceinline__ float funkey(unsigned k) {
    return __uint_as_float((k & 0x80000000u) ? (k ^ 0x80000000u) : ~k);
}

// ---------------- utility fills ----------------
__global__ void fill_f32(float* p, float v, int n) {
    int i = blockIdx.x * blockDim.x + threadIdx.x;
    if (i < n) p[i] = v;
}
__global__ void fill_u32(unsigned* p, unsigned v, int n) {
    int i = blockIdx.x * blockDim.x + threadIdx.x;
    if (i < n) p[i] = v;
}

// ---------------- self-loop attr = mean of incoming edge attrs ----------------
__global__ void loop_sum_k(const int* __restrict__ dst, const float* __restrict__ ea,
                           float* __restrict__ sum, float* __restrict__ cnt, int E) {
    int e = blockIdx.x * blockDim.x + threadIdx.x;
    if (e >= E) return;
    int d = dst[e];
    atomicAdd(&sum[d * 3 + 0], ea[(size_t)e * 3 + 0]);
    atomicAdd(&sum[d * 3 + 1], ea[(size_t)e * 3 + 1]);
    atomicAdd(&sum[d * 3 + 2], ea[(size_t)e * 3 + 2]);
    atomicAdd(&cnt[d], 1.0f);
}
__global__ void loop_div_k(const float* __restrict__ sum, const float* __restrict__ cnt,
                           float* __restrict__ loopa, int N) {
    int i = blockIdx.x * blockDim.x + threadIdx.x;
    if (i >= N) return;
    float c = cnt[i];
    float inv = c > 0.f ? 1.f / c : 0.f;
    loopa[i * 3 + 0] = sum[i * 3 + 0] * inv;
    loopa[i * 3 + 1] = sum[i * 3 + 1] * inv;
    loopa[i * 3 + 2] = sum[i * 3 + 2] * inv;
}

// ---------------- WMMA GEMM: Y[nrows x OC] = X[nrows x K] @ W^T + bias ----------------
// W is row-major [OC x K] (so B = W^T). One wave computes one 16x16 tile of Y.
// K, OC compile-time -> fully unrolled WMMA chain + straight-line stores.
// Requires nrows % 16 == 0 (N = 40000 = 2500*16).
template <int K, int OC>
__global__ void gemm_wmma(const float* __restrict__ X, const float* __restrict__ W,
                          const float* __restrict__ bias, float* __restrict__ Y,
                          int nrows) {
    int wave = blockIdx.x * (blockDim.x >> 5) + (threadIdx.x >> 5);
    constexpr int tilesC = OC >> 4;
    int tilesR = nrows >> 4;
    if (wave >= tilesR * tilesC) return;       // wave-uniform exit: EXEC all-1s at every WMMA
    int tileR = wave / tilesC;
    int tileC = wave - tileR * tilesC;

    int lane = threadIdx.x & 31;
    int half = lane >> 4;        // 0: lanes 0-15, 1: lanes 16-31
    int m    = lane & 15;

    const float* xrow = X + (size_t)(tileR * 16 + m) * K;     // A row (M = m)
    const float* wrow = W + (size_t)(tileC * 16 + m) * K;     // B column n = m (B[k][n] = W[n][k])

    f32x8 acc = {0.f, 0.f, 0.f, 0.f, 0.f, 0.f, 0.f, 0.f};

#pragma unroll
    for (int kk = 0; kk < K; kk += 32) {
        if (kk + 32 < K) __builtin_prefetch(xrow + kk + 32, 0, 0);  // -> global_prefetch_b8
        // ---- A fragment: 16x32 bf16. elems 0-7: K=kk+half*8+j ; elems 8-15: K=kk+16+half*8+j
        const float4* pa = (const float4*)(xrow + kk + half * 8);
        const float4* pq = (const float4*)(xrow + kk + 16 + half * 8);
        float4 a0 = pa[0], a1 = pa[1], a2 = pq[0], a3 = pq[1];
        bf16x16 a;
        a[0]=(__bf16)a0.x; a[1]=(__bf16)a0.y; a[2]=(__bf16)a0.z; a[3]=(__bf16)a0.w;
        a[4]=(__bf16)a1.x; a[5]=(__bf16)a1.y; a[6]=(__bf16)a1.z; a[7]=(__bf16)a1.w;
        a[8]=(__bf16)a2.x; a[9]=(__bf16)a2.y; a[10]=(__bf16)a2.z; a[11]=(__bf16)a2.w;
        a[12]=(__bf16)a3.x; a[13]=(__bf16)a3.y; a[14]=(__bf16)a3.z; a[15]=(__bf16)a3.w;
        // ---- B fragment: 32x16 bf16. elems 0-7: K=kk+half*8+j ; elems 8-15: K=kk+16+half*8+j, column n = lane&15
        const float4* pb = (const float4*)(wrow + kk + half * 8);
        const float4* pbq = (const float4*)(wrow + kk + 16 + half * 8);
        float4 b0 = pb[0], b1 = pb[1], b2 = pbq[0], b3 = pbq[1];
        bf16x16 b;
        b[0]=(__bf16)b0.x; b[1]=(__bf16)b0.y; b[2]=(__bf16)b0.z; b[3]=(__bf16)b0.w;
        b[4]=(__bf16)b1.x; b[5]=(__bf16)b1.y; b[6]=(__bf16)b1.z; b[7]=(__bf16)b1.w;
        b[8]=(__bf16)b2.x; b[9]=(__bf16)b2.y; b[10]=(__bf16)b2.z; b[11]=(__bf16)b2.w;
        b[12]=(__bf16)b3.x; b[13]=(__bf16)b3.y; b[14]=(__bf16)b3.z; b[15]=(__bf16)b3.w;

        acc = __builtin_amdgcn_wmma_f32_16x16x32_bf16(false, a, false, b,
                                                      (short)0, acc, false, false);
    }

    // ---- D layout: VGPR v -> row (v + 8*half), col = lane&15. Full tiles: no guards.
    int col  = tileC * 16 + m;
    float bv = bias[col];
    float* yb = Y + (size_t)(tileR * 16 + half * 8) * OC + col;
#pragma unroll
    for (int v = 0; v < 8; ++v)
        yb[(size_t)v * OC] = acc[v] + bv;
}

// ---------------- attention logits + segment max (per edge*head thread) ----------------
__global__ void alpha_k(const float* __restrict__ xl, const float* __restrict__ xr,
                        const int* __restrict__ src, const int* __restrict__ dst,
                        const float* __restrict__ eattr, const float* __restrict__ loopa,
                        const float* __restrict__ We, const float* __restrict__ att,
                        float* __restrict__ alpha, unsigned* __restrict__ amax,
                        int E, int ETOT) {
    __shared__ float sWe[HC * EDIM];
    __shared__ float sAtt[HC];
    for (int i = threadIdx.x; i < HC * EDIM; i += blockDim.x) sWe[i] = We[i];
    for (int i = threadIdx.x; i < HC; i += blockDim.x) sAtt[i] = att[i];
    __syncthreads();

    int t = blockIdx.x * blockDim.x + threadIdx.x;
    if (t >= ETOT * HEADS) return;
    int e = t >> 3, h = t & 7;
    int s, d; const float* ea;
    if (e < E) { s = src[e]; d = dst[e]; ea = eattr + (size_t)e * EDIM; }
    else       { s = d = e - E;          ea = loopa + (size_t)(e - E) * EDIM; }
    float a0 = ea[0], a1 = ea[1], a2 = ea[2];

    const float* pl = xl + (size_t)s * HC + h * CPH;
    const float* pr = xr + (size_t)d * HC + h * CPH;
    float acc = 0.f;
#pragma unroll
    for (int c = 0; c < CPH; ++c) {
        int j = h * CPH + c;
        float eemb = fmaf(sWe[j * 3], a0, fmaf(sWe[j * 3 + 1], a1, sWe[j * 3 + 2] * a2));
        float mv = pl[c] + pr[c] + eemb;
        mv = mv > 0.f ? mv : NSLOPE * mv;
        acc = fmaf(mv, sAtt[j], acc);
    }
    alpha[t] = acc;
    atomicMax(&amax[d * HEADS + h], fkey(acc));
}

// ---------------- exp(alpha - max) + segment sum ----------------
__global__ void expsum_k(const int* __restrict__ dst, float* __restrict__ alpha,
                         const unsigned* __restrict__ amax, float* __restrict__ denom,
                         int E, int ETOT) {
    int t = blockIdx.x * blockDim.x + threadIdx.x;
    if (t >= ETOT * HEADS) return;
    int e = t >> 3, h = t & 7;
    int d = (e < E) ? dst[e] : (e - E);
    float ex = expf(alpha[t] - funkey(amax[d * HEADS + h]));
    alpha[t] = ex;
    atomicAdd(&denom[d * HEADS + h], ex);
}

// ---------------- weighted scatter-add aggregation ----------------
__global__ void agg_k(const int* __restrict__ src, const int* __restrict__ dst,
                      const float* __restrict__ alpha, const float* __restrict__ denom,
                      const float* __restrict__ xl, float* __restrict__ agg,
                      int E, int ETOT) {
    int t = blockIdx.x * blockDim.x + threadIdx.x;
    if (t >= ETOT * HEADS) return;
    int e = t >> 3, h = t & 7;
    int s, d;
    if (e < E) { s = src[e]; d = dst[e]; } else { s = d = e - E; }
    float a = alpha[t] / (denom[d * HEADS + h] + 1e-16f);
    const float* pl = xl + (size_t)s * HC + h * CPH;
    float* po = agg + (size_t)d * HC + h * CPH;
#pragma unroll
    for (int c = 0; c < CPH; ++c) atomicAdd(&po[c], pl[c] * a);
}

// ---------------- bias + ELU ----------------
__global__ void bias_elu_k(const float* __restrict__ agg, const float* __restrict__ bias,
                           float* __restrict__ out, int n) {
    int t = blockIdx.x * blockDim.x + threadIdx.x;
    if (t >= n) return;
    float v = agg[t] + bias[t & (HC - 1)];
    out[t] = v > 0.f ? v : (expf(v) - 1.f);
}

// ---------------- host-side layer driver ----------------
template <int KIN>
static void run_layer(const float* xin,
                      const float* Wl, const float* bl, const float* Wr, const float* br,
                      const float* We, const float* att, const float* bias,
                      const int* src, const int* dst, const float* eattr, const float* loopa,
                      float* xl, float* xr, float* alpha, unsigned* amax, float* denom,
                      float* agg, float* hout, int N, int E, hipStream_t stream) {
    int ETOT = E + N;
    int waves = (N / 16) * (HC / 16);
    int gb = (waves + 7) / 8;
    gemm_wmma<KIN, HC><<<gb, 256, 0, stream>>>(xin, Wl, bl, xl, N);
    gemm_wmma<KIN, HC><<<gb, 256, 0, stream>>>(xin, Wr, br, xr, N);
    fill_u32<<<(N * HEADS + 255) / 256, 256, 0, stream>>>(amax, 0u, N * HEADS);
    fill_f32<<<(N * HEADS + 255) / 256, 256, 0, stream>>>(denom, 0.f, N * HEADS);
    fill_f32<<<(N * HC + 255) / 256, 256, 0, stream>>>(agg, 0.f, N * HC);
    int tt = ETOT * HEADS;
    alpha_k <<<(tt + 255) / 256, 256, 0, stream>>>(xl, xr, src, dst, eattr, loopa, We, att, alpha, amax, E, ETOT);
    expsum_k<<<(tt + 255) / 256, 256, 0, stream>>>(dst, alpha, amax, denom, E, ETOT);
    agg_k   <<<(tt + 255) / 256, 256, 0, stream>>>(src, dst, alpha, denom, xl, agg, E, ETOT);
    bias_elu_k<<<(N * HC + 255) / 256, 256, 0, stream>>>(agg, bias, hout, N * HC);
}

extern "C" void kernel_launch(void* const* d_in, const int* in_sizes, int n_in,
                              void* d_out, int out_size, void* d_ws, size_t ws_size,
                              hipStream_t stream) {
    const float* x     = (const float*)d_in[0];
    const int*   esrc  = (const int*)  d_in[1];
    const int*   edst  = (const int*)  d_in[2];
    const float* eattr = (const float*)d_in[3];
    const float* Wl1 = (const float*)d_in[4];  const float* bl1 = (const float*)d_in[5];
    const float* Wr1 = (const float*)d_in[6];  const float* br1 = (const float*)d_in[7];
    const float* We1 = (const float*)d_in[8];  const float* att1= (const float*)d_in[9];
    const float* b1  = (const float*)d_in[10];
    const float* Wl2 = (const float*)d_in[11]; const float* bl2 = (const float*)d_in[12];
    const float* Wr2 = (const float*)d_in[13]; const float* br2 = (const float*)d_in[14];
    const float* We2 = (const float*)d_in[15]; const float* att2= (const float*)d_in[16];
    const float* b2  = (const float*)d_in[17];
    const float* Wf  = (const float*)d_in[18]; const float* bf  = (const float*)d_in[19];

    const int N = in_sizes[0] / IN_CH;
    const int E = in_sizes[1];
    const int ETOT = E + N;

    // ---- workspace carve-up (floats) ----
    float* w = (float*)d_ws;
    size_t off = 0;
    float* loop_sum  = w + off; off += (size_t)N * 3;
    float* loop_cnt  = w + off; off += (size_t)N;
    float* loop_attr = w + off; off += (size_t)N * 3;
    float* xl        = w + off; off += (size_t)N * HC;
    float* xr        = w + off; off += (size_t)N * HC;
    float* alpha     = w + off; off += (size_t)ETOT * HEADS;
    float* denom     = w + off; off += (size_t)N * HEADS;
    float* agg       = w + off; off += (size_t)N * HC;
    float* hbuf      = w + off; off += (size_t)N * HC;
    unsigned* amax   = (unsigned*)(w + off); off += (size_t)N * HEADS;

    // ---- self-loop attributes (mean of incoming) ----
    fill_f32<<<(N * 4 + 255) / 256, 256, 0, stream>>>(loop_sum, 0.f, N * 4); // sums + cnt contiguous
    loop_sum_k<<<(E + 255) / 256, 256, 0, stream>>>(edst, eattr, loop_sum, loop_cnt, E);
    loop_div_k<<<(N + 255) / 256, 256, 0, stream>>>(loop_sum, loop_cnt, loop_attr, N);

    // ---- layer 1 (input dim 64) -> hbuf ----
    run_layer<IN_CH>(x, Wl1, bl1, Wr1, br1, We1, att1, b1,
                     esrc, edst, eattr, loop_attr,
                     xl, xr, alpha, amax, denom, agg, hbuf, N, E, stream);
    // ---- layer 2 (input dim 128) -> hbuf (in-place safe: input consumed by GEMMs first) ----
    run_layer<HC>(hbuf, Wl2, bl2, Wr2, br2, We2, att2, b2,
                  esrc, edst, eattr, loop_attr,
                  xl, xr, alpha, amax, denom, agg, hbuf, N, E, stream);

    // ---- final linear: d_out = hbuf @ Wf^T + bf ----
    int waves = (N / 16) * (OUTC / 16);
    gemm_wmma<HC, OUTC><<<(waves + 7) / 8, 256, 0, stream>>>(hbuf, Wf, bf, (float*)d_out, N);
}
